// RotatEPredictor_69965017252750
// MI455X (gfx1250) — matrix operand order repaired and checked by reading, not compiled, not run on hardware
//
#include <hip/hip_runtime.h>

// CDNA5 / gfx1250: wave32, WMMA 16x16x32 f16 -> f32 accumulate.
typedef __attribute__((ext_vector_type(16))) _Float16 v16h;
typedef __attribute__((ext_vector_type(8)))  float    v8f;

#define D    128
#define TILE 16

// ---------------------------------------------------------------------------
// Pack W_w (D x D fp32 row-major; B-matrix element B[k][n] = W_w[n][k]) into
// f16 B-fragment order: index = ((colTile*4 + t)*32 + lane)*16 + e
// B layout (16-bit, 32x16): lane holds N = lane&15, K = (e/8)*16 + (lane>>4)*8 + (e%8).
// ---------------------------------------------------------------------------
__global__ void pack_w_kernel(const float* __restrict__ W, _Float16* __restrict__ wB) {
  int i = blockIdx.x * blockDim.x + threadIdx.x;   // [0, 8*4*32*16) = 16384
  int e = i & 15;
  int l = (i >> 4) & 31;
  int t = (i >> 9) & 3;
  int j = i >> 11;
  int n = j * TILE + (l & 15);
  int k = t * 32 + ((e >> 3) << 4) + ((l >> 4) << 3) + (e & 7);
  wB[i] = (_Float16)W[n * D + k];
}

// ---------------------------------------------------------------------------
// Pack h (N x D fp32) into f16 A-fragment order:
//   index = ((rowTile*4 + t)*32 + lane)*16 + e
// A layout (16-bit, 16x32): lane holds M = lane&15,
//   K(e) = (e/8)*16 + (lane>>4)*8 + (e%8)  -> two contiguous 8-float runs.
// Each thread packs one (rowTile, t, lane) fragment slice: 4x float4 loads,
// one 32-byte contiguous v16h store.
// ---------------------------------------------------------------------------
__global__ void pack_h_kernel(const float* __restrict__ h, _Float16* __restrict__ hA,
                              int rowTiles) {
  int i = blockIdx.x * blockDim.x + threadIdx.x;   // [0, rowTiles*4*32)
  if (i >= rowTiles * 128) return;
  int l = i & 31;
  int t = (i >> 5) & 3;
  int r = i >> 7;
  int m  = r * TILE + (l & 15);
  int k0 = t * 32 + ((l >> 4) << 3);               // lane_hi * 8
  const float4* p = (const float4*)(h + (size_t)m * D + k0);
  float4 f0 = p[0];                                 // K = k0 .. k0+3
  float4 f1 = p[1];                                 // K = k0+4 .. k0+7
  const float4* q = (const float4*)(h + (size_t)m * D + k0 + 16);
  float4 f2 = q[0];                                 // K = k0+16 .. k0+19
  float4 f3 = q[1];                                 // K = k0+20 .. k0+23
  v16h a;
  a[0]  = (_Float16)f0.x; a[1]  = (_Float16)f0.y;
  a[2]  = (_Float16)f0.z; a[3]  = (_Float16)f0.w;
  a[4]  = (_Float16)f1.x; a[5]  = (_Float16)f1.y;
  a[6]  = (_Float16)f1.z; a[7]  = (_Float16)f1.w;
  a[8]  = (_Float16)f2.x; a[9]  = (_Float16)f2.y;
  a[10] = (_Float16)f2.z; a[11] = (_Float16)f2.w;
  a[12] = (_Float16)f3.x; a[13] = (_Float16)f3.y;
  a[14] = (_Float16)f3.z; a[15] = (_Float16)f3.w;
  *(v16h*)(hA + (size_t)i * 16) = a;
}

// ---------------------------------------------------------------------------
// Wh = h @ W_w^T + b via V_WMMA_F32_16X16X32_F16.
// One block (8 wave32) per 16-row tile; wave w owns output column tile w.
// All 8 waves reuse the same A fragments (L1 hits). Bias is folded into the
// initial f32 accumulator. K=128 -> 4 chained WMMAs.
// C/D layout: VGPR g, lane -> M = g + 8*(lane>>4), N = lane&15.
// ---------------------------------------------------------------------------
__global__ void gemm_kernel(const _Float16* __restrict__ hA,
                            const _Float16* __restrict__ wB,
                            const float* __restrict__ bias,
                            float* __restrict__ Wh) {
  int r    = blockIdx.x;
  int wv   = threadIdx.x >> 5;   // 0..7 : column tile
  int lane = threadIdx.x & 31;

  float bval = bias[wv * TILE + (lane & 15)];
  v8f c;
#pragma unroll
  for (int g = 0; g < 8; ++g) c[g] = bval;

#pragma unroll
  for (int t = 0; t < 4; ++t) {
    v16h a = *(const v16h*)(hA + (size_t)((r * 4 + t) * 32 + lane) * 16);
    v16h b = *(const v16h*)(wB + (size_t)((wv * 4 + t) * 32 + lane) * 16);
    // 8 args: (neg_a, A, neg_b, B, c_mod, C, reuse_a, reuse_b)
    c = __builtin_amdgcn_wmma_f32_16x16x32_f16(false, a, false, b,
                                               (short)0, c, false, false);
  }

  int row = r * TILE + ((lane >> 4) << 3);
  int col = wv * TILE + (lane & 15);
#pragma unroll
  for (int g = 0; g < 8; ++g)
    Wh[(size_t)(row + g) * D + col] = c[g];
}

// ---------------------------------------------------------------------------
// score[e] = dot(Wh[src[e]], h[dst[e]]).  8 lanes per edge, 4x float4 per
// lane (contiguous 64B per lane; the 8-lane group covers a 512B row), shfl_xor reduction.
// Wh + h both fit in the 192MB L2 -> gathers are L2-resident.
// ---------------------------------------------------------------------------
__global__ void edge_kernel(const float* __restrict__ Wh,
                            const float* __restrict__ h,
                            const int* __restrict__ src,
                            const int* __restrict__ dst,
                            float* __restrict__ out, int E) {
  int tid  = blockIdx.x * blockDim.x + threadIdx.x;
  int edge = tid >> 3;
  int sub  = tid & 7;
  if (edge >= E) return;
  size_t s = (size_t)src[edge];
  size_t d = (size_t)dst[edge];
  const float4* ps = (const float4*)(Wh + s * D + sub * 16);
  const float4* pd = (const float4*)(h  + d * D + sub * 16);
  float acc = 0.f;
#pragma unroll
  for (int q = 0; q < 4; ++q) {
    float4 a = ps[q];
    float4 b = pd[q];
    acc += a.x * b.x + a.y * b.y + a.z * b.z + a.w * b.w;
  }
  acc += __shfl_xor(acc, 4, 8);
  acc += __shfl_xor(acc, 2, 8);
  acc += __shfl_xor(acc, 1, 8);
  if (sub == 0) out[edge] = acc;
}

extern "C" void kernel_launch(void* const* d_in, const int* in_sizes, int n_in,
                              void* d_out, int out_size, void* d_ws, size_t ws_size,
                              hipStream_t stream) {
  const float* h   = (const float*)d_in[0];
  const int*   src = (const int*)d_in[1];   // integer input -> const int* per harness
  const int*   dst = (const int*)d_in[2];
  const float* W_w = (const float*)d_in[3];
  const float* W_b = (const float*)d_in[4];

  int N        = in_sizes[0] / D;           // 100000
  int E        = in_sizes[1];               // 600000
  int rowTiles = N / TILE;                  // 6250 (exact)

  char*     ws = (char*)d_ws;
  float*    Wh = (float*)ws;                                             // N*D f32
  _Float16* hA = (_Float16*)(ws + (size_t)N * D * sizeof(float));        // N*D f16
  _Float16* wB = (_Float16*)(ws + (size_t)N * D * (sizeof(float) + 2));  // D*D f16

  pack_w_kernel<<<(8 * 4 * 32 * 16) / 256, 256, 0, stream>>>(W_w, wB);
  pack_h_kernel<<<(rowTiles * 128 + 255) / 256, 256, 0, stream>>>(h, hA, rowTiles);
  gemm_kernel<<<rowTiles, 256, 0, stream>>>(hA, wB, W_b, Wh);
  edge_kernel<<<(E * 8 + 255) / 256, 256, 0, stream>>>(Wh, h, src, dst,
                                                       (float*)d_out, E);
}
